// Upsample_30777735643621
// MI455X (gfx1250) — matrix operand order repaired and checked by reading, not compiled, not run on hardware
//
#include <hip/hip_runtime.h>

typedef __attribute__((ext_vector_type(16))) __bf16 v16bf;
typedef __attribute__((ext_vector_type(8)))  float  v8f;
typedef int vi4 __attribute__((vector_size(16)));

#define BATCH 16
#define HIN   64
#define WIN   64
#define CH    128
#define OCH   128
#define HOUT  128
#define WOUT  128

#define ROWS_PER_BLOCK 4
#define LDS_ROWS 6            // rows i0-1 .. i0+4
#define LDS_COLS 66           // cols -1 .. 64

#if __has_builtin(__builtin_amdgcn_global_load_async_to_lds_b128)
#define HAVE_ASYNC_LDS 1
#else
#define HAVE_ASYNC_LDS 0
#endif

__device__ __forceinline__ unsigned short f2bf(float f) {
    unsigned int u = __float_as_uint(f);
    u += 0x7fffu + ((u >> 16) & 1u);      // round-to-nearest-even
    return (unsigned short)(u >> 16);
}

// ---------------- prep 1: x fp32 -> bf16 ----------------
__global__ void cvt_x_bf16(const float* __restrict__ x,
                           unsigned int* __restrict__ xb, int n2) {
    int i = blockIdx.x * blockDim.x + threadIdx.x;
    if (i < n2) {
        float2 f = ((const float2*)x)[i];
        xb[i] = (unsigned int)f2bf(f.x) | ((unsigned int)f2bf(f.y) << 16);
    }
}

// ---------------- prep 2: fold FIR into 4 phase kernels ----------------
// A[py][dy][k]: FIR/conv fusion coefficients (fir1 = [.25,.75,.75,.25])
__constant__ float A_TAB[2][3][3] = {
    {{0.00f, 0.00f, 0.25f}, {0.25f, 0.75f, 0.75f}, {0.75f, 0.25f, 0.00f}},
    {{0.00f, 0.25f, 0.75f}, {0.75f, 0.75f, 0.25f}, {0.25f, 0.00f, 0.00f}}};

// out layout: kp[(phase*9 + tap)*OCH + o]*CH + c   (c contiguous => B frags contiguous)
__global__ void build_kp(const float* __restrict__ w,
                         unsigned short* __restrict__ kp) {
    int idx = blockIdx.x * blockDim.x + threadIdx.x;
    if (idx >= 4 * 9 * CH * OCH) return;
    int c  = idx & 127;
    int o  = (idx >> 7) & 127;
    int pt = idx >> 14;            // phase*9 + tap
    int tap = pt % 9, phase = pt / 9;
    int dy = tap / 3, dx = tap % 3;
    int py = phase >> 1, px = phase & 1;
    float acc = 0.f;
    for (int k = 0; k < 3; ++k) {
        float ay = A_TAB[py][dy][k];
        if (ay == 0.f) continue;
        for (int l = 0; l < 3; ++l) {
            float ax = A_TAB[px][dx][l];
            if (ax == 0.f) continue;
            acc += ay * ax * w[((k * 3 + l) * CH + c) * OCH + o];
        }
    }
    kp[((size_t)pt * OCH + o) * CH + c] = f2bf(acc);
}

// ---------------- main: implicit-GEMM subpixel conv via WMMA bf16 ----------------
union Frag { uint4 q[2]; v16bf v; };

__launch_bounds__(256)
__global__ void upconv_wmma(const unsigned short* __restrict__ xb,
                            const unsigned short* __restrict__ kp,
                            const float* __restrict__ bias,
                            float* __restrict__ out) {
    __shared__ __align__(16) unsigned short lds[LDS_ROWS * LDS_COLS * CH]; // 101376 B

    const int bx    = blockIdx.x;
    const int phase = bx >> 8;                 // 256 blocks per phase
    const int rem   = bx & 255;
    const int bb    = rem >> 4;                // batch
    const int i0    = (rem & 15) * ROWS_PER_BLOCK;  // first of 4 source rows
    const int py    = phase >> 1, px = phase & 1;

    // ---- halo-tile fill (rows i0-1..i0+4, cols -1..64, bf16), async -> LDS ----
    const int tid = threadIdx.x;
    for (int ch = tid; ch < LDS_ROWS * LDS_COLS * (CH / 8); ch += 256) {
        int cpart = ch & 15;                   // 8-ushort (16 B) chunk in channel dim
        int rc    = ch >> 4;
        int col   = rc % LDS_COLS;
        int r     = rc / LDS_COLS;
        int xr = i0 + r - 1;
        int xc = col - 1;
        unsigned short* dst = &lds[(r * LDS_COLS + col) * CH + cpart * 8];
        if ((unsigned)xr < HIN && (unsigned)xc < WIN) {
            const unsigned short* src =
                &xb[(((size_t)bb * HIN + xr) * WIN + xc) * CH + cpart * 8];
#if HAVE_ASYNC_LDS
            __builtin_amdgcn_global_load_async_to_lds_b128(
                (__attribute__((address_space(1))) vi4*)src,
                (__attribute__((address_space(3))) vi4*)dst, 0, 0);
#else
            *(uint4*)dst = *(const uint4*)src;
#endif
        } else {
            *(uint4*)dst = make_uint4(0u, 0u, 0u, 0u);   // zero-pad halo
        }
    }
#if HAVE_ASYNC_LDS
#if __has_builtin(__builtin_amdgcn_s_wait_asynccnt)
    __builtin_amdgcn_s_wait_asynccnt(0);
#else
    asm volatile("s_wait_asynccnt 0x0" ::: "memory");
#endif
#endif
    __syncthreads();

    const int lane = tid & 31;
    const int wave = tid >> 5;
    const int wm   = wave & 3;           // 16-wide j' subtile
    const int wn   = wave >> 2;          // 64-wide o half
    const int half = (lane >> 4) & 1;
    const int mrow = lane & 15;

    v8f zero = {};
    v8f acc[ROWS_PER_BLOCK][4];
#pragma unroll
    for (int r2 = 0; r2 < ROWS_PER_BLOCK; ++r2)
#pragma unroll
        for (int nt = 0; nt < 4; ++nt) acc[r2][nt] = zero;

    const unsigned short* kpp = kp + (size_t)phase * 9 * OCH * CH;

    for (int tap = 0; tap < 9; ++tap) {
        const int dy = tap / 3, dx = tap % 3;
        const unsigned short* kt = kpp + (size_t)tap * OCH * CH;
        // prefetch next tap's B region (speculative; harmless at the tail)
        __builtin_prefetch(kt + (size_t)OCH * CH + (size_t)(wn * 64 + mrow) * CH, 0, 1);
#pragma unroll
        for (int c0 = 0; c0 < CH; c0 += 32) {
            // B fragments: lane holds K = c0 + 16*half + e (contiguous 32B)
            Frag bfr[4];
#pragma unroll
            for (int nt = 0; nt < 4; ++nt) {
                const unsigned short* p =
                    kt + (size_t)(wn * 64 + nt * 16 + mrow) * CH + c0 + 16 * half;
                bfr[nt].q[0] = *(const uint4*)p;
                bfr[nt].q[1] = *(const uint4*)(p + 8);
            }
#pragma unroll
            for (int r2 = 0; r2 < ROWS_PER_BLOCK; ++r2) {
                // A fragment: lane holds K = c0 + 16*half + {0..15}
                const int ldsrow = r2 + 2 - dy;
                const int ldscol = wm * 16 + mrow + 2 - dx;
                const unsigned short* ap =
                    &lds[(ldsrow * LDS_COLS + ldscol) * CH + c0 + 16 * half];
                Frag af;
                af.q[0] = *(const uint4*)ap;
                af.q[1] = *(const uint4*)(ap + 8);
#pragma unroll
                for (int nt = 0; nt < 4; ++nt) {
                    acc[r2][nt] = __builtin_amdgcn_wmma_f32_16x16x32_bf16(
                        false, af.v, false, bfr[nt].v,
                        (short)0, acc[r2][nt], false, false);
                }
            }
        }
    }

    // ---- epilogue: +bias, store interleaved phase output ----
    float bv[4];
#pragma unroll
    for (int nt = 0; nt < 4; ++nt) bv[nt] = bias[wn * 64 + nt * 16 + mrow];

#pragma unroll
    for (int r2 = 0; r2 < ROWS_PER_BLOCK; ++r2) {
        const int m = ((i0 + r2) << 1) + py;
#pragma unroll
        for (int nt = 0; nt < 4; ++nt) {
            const int o = wn * 64 + nt * 16 + mrow;
#pragma unroll
            for (int v = 0; v < 8; ++v) {
                const int jp = wm * 16 + v + 8 * half;   // D layout: M = v + 8*(lane/16)
                const int n  = (jp << 1) + px;
                out[(((size_t)bb * HOUT + m) * WOUT + n) * OCH + o] =
                    acc[r2][nt][v] + bv[nt];
            }
        }
    }
}

extern "C" void kernel_launch(void* const* d_in, const int* in_sizes, int n_in,
                              void* d_out, int out_size, void* d_ws, size_t ws_size,
                              hipStream_t stream) {
    const float* x = (const float*)d_in[0];  // (16,64,64,128)
    const float* w = (const float*)d_in[1];  // (3,3,128,128)
    const float* b = (const float*)d_in[2];  // (128,)
    float* out = (float*)d_out;              // (16,128,128,128)

    unsigned short* xb  = (unsigned short*)d_ws;                 // 16 MB bf16 x
    unsigned short* kpb = xb + (size_t)BATCH * HIN * WIN * CH;   // 1.15 MB bf16 Kp

    const int n2 = BATCH * HIN * WIN * CH / 2;
    cvt_x_bf16<<<(n2 + 255) / 256, 256, 0, stream>>>(x, (unsigned int*)xb, n2);

    const int nk = 4 * 9 * CH * OCH;
    build_kp<<<(nk + 255) / 256, 256, 0, stream>>>(w, kpb);

    upconv_wmma<<<4 * BATCH * (HIN / ROWS_PER_BLOCK), 256, 0, stream>>>(xb, kpb, b, out);
}
